// FactorGATPolicy_35553739276560
// MI455X (gfx1250) — matrix operand order repaired and checked by reading, not compiled, not run on hardware
//
#include <hip/hip_runtime.h>
#include <hip/hip_bf16.h>
#include <math.h>

typedef __attribute__((ext_vector_type(16))) _Float16 v16h;
typedef __attribute__((ext_vector_type(8)))  _Float16 v8h;
typedef __attribute__((ext_vector_type(8)))  float    v8f;

#define LN_EPS 1e-5f
#define NEG_SLOPE 0.2f

__device__ __forceinline__ float warpSumF(float v) {
  #pragma unroll
  for (int off = 16; off > 0; off >>= 1) v += __shfl_xor(v, off, 32);
  return v;
}

__device__ __forceinline__ float leakyF(float x) { return x > 0.f ? x : NEG_SLOPE * x; }

__device__ __forceinline__ void atomicMaxF(float* addr, float val) {
  if (val >= 0.f) atomicMax((int*)addr, __float_as_int(val));
  else            atomicMin((unsigned int*)addr, __float_as_uint(val));
}

// ---------------- regime layernorm (16 elements) ----------------
__global__ void k_reg_ln(const float* __restrict__ r, const float* __restrict__ g,
                         const float* __restrict__ b, float* __restrict__ out, int n) {
  if (blockIdx.x == 0 && threadIdx.x == 0) {
    float mu = 0.f;
    for (int i = 0; i < n; ++i) mu += r[i];
    mu /= n;
    float var = 0.f;
    for (int i = 0; i < n; ++i) { float d = r[i] - mu; var += d * d; }
    var /= n;
    float rs = rsqrtf(var + LN_EPS);
    for (int i = 0; i < n; ++i) out[i] = (r[i] - mu) * rs * g[i] + b[i];
  }
}

// ------------- node layernorm + concat regime, emit f16 [N,160] -------------
__global__ void k_node_ln(const float* __restrict__ x, const float* __restrict__ gamma,
                          const float* __restrict__ beta, const float* __restrict__ rln,
                          _Float16* __restrict__ h16, int N) {
  int wave = (int)((blockIdx.x * blockDim.x + threadIdx.x) >> 5);
  int lane = threadIdx.x & 31;
  if (wave >= N) return;
  const float* row = x + (size_t)wave * 128;
  float s1 = 0.f, s2 = 0.f;
  for (int k = lane; k < 128; k += 32) { float v = row[k]; s1 += v; s2 += v * v; }
  s1 = warpSumF(s1); s2 = warpSumF(s2);
  float mu  = s1 * (1.f / 128.f);
  float var = s2 * (1.f / 128.f) - mu * mu;
  float rs  = rsqrtf(var + LN_EPS);
  _Float16* dst = h16 + (size_t)wave * 160;
  for (int k = lane; k < 160; k += 32) {
    float v;
    if (k < 128)      v = (row[k] - mu) * rs * gamma[k] + beta[k];
    else if (k < 144) v = rln[k - 128];
    else              v = 0.f;
    dst[k] = (_Float16)v;
  }
}

// ------------- pack row-major W[K,Ncols] f32 -> WMMA B-fragment order f16 -------------
// layout: Wp[(((kt*NT)+nt)*32 + lane)*16 + j]; lanes 0-15: K=kt*32+j, lanes 16-31: K=kt*32+16+j
__global__ void k_pack_w(const float* __restrict__ W, _Float16* __restrict__ Wp,
                         int K, int KT, int NT, int Ncols) {
  int idx = blockIdx.x * blockDim.x + threadIdx.x;
  int total = KT * NT * 32 * 16;
  if (idx >= total) return;
  int j    = idx & 15;
  int lane = (idx >> 4) & 31;
  int nt   = (idx >> 9) % NT;
  int kt   = idx / (NT * 512);
  int n    = nt * 16 + (lane & 15);
  int k    = kt * 32 + ((lane < 16) ? j : 16 + j);
  float v  = (k < K) ? W[(size_t)k * Ncols + n] : 0.f;
  Wp[idx]  = (_Float16)v;
}

// ------------- WMMA GEMM: A[f16 row-major, lda] x Wp(packed) -> C[f32, ldc] -------------
__global__ void k_gemm_wmma(const _Float16* __restrict__ A, const _Float16* __restrict__ Wp,
                            float* __restrict__ C, int nRowTiles, int nrows,
                            int lda, int KT, int NT, int ldc) {
  int wid  = (int)((blockIdx.x * blockDim.x + threadIdx.x) >> 5);
  int lane = threadIdx.x & 31;
  int rt = wid / NT, ct = wid % NT;
  if (rt >= nRowTiles) return;           // wave-uniform: EXEC all-1s at WMMA
  int row   = rt * 16 + (lane & 15);
  int kbase = (lane >> 4) ? 16 : 0;      // A-fragment K sub-base per lane group
  v8f acc = {};
  for (int kt = 0; kt < KT; ++kt) {
    union { v16h v; v8h h[2]; } a, b;
    const _Float16* ap = A + (size_t)row * lda + kt * 32 + kbase;
    a.h[0] = *(const v8h*)(ap);
    a.h[1] = *(const v8h*)(ap + 8);
    const _Float16* bp = Wp + ((size_t)(kt * NT + ct) * 32 + lane) * 16;
    b.h[0] = *(const v8h*)(bp);
    b.h[1] = *(const v8h*)(bp + 8);
    if (kt + 1 < KT)
      __builtin_prefetch(ap + 32, 0, 1);   // global_prefetch_b8 of next K chunk
    acc = __builtin_amdgcn_wmma_f32_16x16x32_f16(false, a.v, false, b.v,
                                                 (short)0, acc, false, false);
  }
  int col   = ct * 16 + (lane & 15);
  int rbase = rt * 16 + ((lane >> 4) ? 8 : 0);
  #pragma unroll
  for (int r = 0; r < 8; ++r) {
    int orow = rbase + r;
    if (orow < nrows) C[(size_t)orow * ldc + col] = acc[r];
  }
}

// ------------- per-(node,head) attention coefficients -------------
__global__ void k_alpha(const float* __restrict__ xw, const float* __restrict__ a_src,
                        const float* __restrict__ a_dst, float* __restrict__ as,
                        float* __restrict__ ad, int N, int H, int Cc) {
  int id = blockIdx.x * blockDim.x + threadIdx.x;
  if (id >= N * H) return;
  int n = id / H, h = id % H;
  const float* xr = xw + (size_t)n * H * Cc + (size_t)h * Cc;
  const float* sv = a_src + h * Cc;
  const float* dv = a_dst + h * Cc;
  float vs = 0.f, vd = 0.f;
  for (int c = 0; c < Cc; ++c) { float v = xr[c]; vs += v * sv[c]; vd += v * dv[c]; }
  as[id] = vs; ad[id] = vd;
}

__global__ void k_fill(float* __restrict__ p, float v, long long n) {
  long long i = (long long)blockIdx.x * blockDim.x + threadIdx.x;
  if (i < n) p[i] = v;
}

__device__ __forceinline__ void edge_sd(const int* __restrict__ esrc,
                                        const int* __restrict__ edst,
                                        int E, long long e, int& s, int& d) {
  if (e < E) { s = esrc[e]; d = edst[e]; }
  else       { s = (int)(e - E); d = s; }      // self loops appended
}

// pass 1: segment max
__global__ void k_edge_max(const int* __restrict__ esrc, const int* __restrict__ edst,
                           int E, int N, int H, const float* __restrict__ as,
                           const float* __restrict__ ad, float* __restrict__ m) {
  long long id = (long long)blockIdx.x * blockDim.x + threadIdx.x;
  long long tot = (long long)(E + N) * H;
  if (id >= tot) return;
  long long e = id / H; int h = (int)(id % H);
  int s, d; edge_sd(esrc, edst, E, e, s, d);
  float v = leakyF(as[(size_t)s * H + h] + ad[(size_t)d * H + h]);
  atomicMaxF(&m[(size_t)d * H + h], v);
}

// pass 2: exp(e - max), store numerator, accumulate denominator
__global__ void k_edge_exp(const int* __restrict__ esrc, const int* __restrict__ edst,
                           int E, int N, int H, const float* __restrict__ as,
                           const float* __restrict__ ad, const float* __restrict__ m,
                           float* __restrict__ denom, float* __restrict__ alpha) {
  long long id = (long long)blockIdx.x * blockDim.x + threadIdx.x;
  long long tot = (long long)(E + N) * H;
  if (id >= tot) return;
  long long e = id / H; int h = (int)(id % H);
  int s, d; edge_sd(esrc, edst, E, e, s, d);
  float v  = leakyF(as[(size_t)s * H + h] + ad[(size_t)d * H + h]);
  float ex = expf(v - m[(size_t)d * H + h]);
  alpha[id] = ex;
  atomicAdd(&denom[(size_t)d * H + h], ex);
}

// pass 3: weighted message scatter-add
__global__ void k_edge_msg(const int* __restrict__ esrc, const int* __restrict__ edst,
                           int E, int N, int H, int Cc, const float* __restrict__ xw,
                           const float* __restrict__ alpha, const float* __restrict__ denom,
                           float* __restrict__ out) {
  long long id = (long long)blockIdx.x * blockDim.x + threadIdx.x;
  int HC = H * Cc;
  long long tot = (long long)(E + N) * HC;
  if (id >= tot) return;
  long long e = id / HC;
  int hc = (int)(id % HC);
  int h  = hc / Cc;
  int s, d; edge_sd(esrc, edst, E, e, s, d);
  float a = alpha[e * H + h] / denom[(size_t)d * H + h];
  atomicAdd(&out[(size_t)d * HC + hc], xw[(size_t)s * HC + hc] * a);
}

// bias + ELU -> f16 (feeds next WMMA GEMM)
__global__ void k_bias_elu_f16(const float* __restrict__ in, const float* __restrict__ bias,
                               _Float16* __restrict__ out, long long total, int F) {
  long long id = (long long)blockIdx.x * blockDim.x + threadIdx.x;
  if (id >= total) return;
  int f = (int)(id % F);
  float v = in[id] + bias[f];
  v = v > 0.f ? v : (expf(v) - 1.f);
  out[id] = (_Float16)v;
}

// bias + ELU in-place f32
__global__ void k_bias_elu_f32(float* __restrict__ io, const float* __restrict__ bias,
                               long long total, int F) {
  long long id = (long long)blockIdx.x * blockDim.x + threadIdx.x;
  if (id >= total) return;
  int f = (int)(id % F);
  float v = io[id] + bias[f];
  io[id] = v > 0.f ? v : (expf(v) - 1.f);
}

// final projection to logits
__global__ void k_logits(const float* __restrict__ h, const float* __restrict__ Wout,
                         const float* __restrict__ bout, float* __restrict__ logits,
                         int N, int F) {
  int n = blockIdx.x * blockDim.x + threadIdx.x;
  if (n >= N) return;
  const float* r = h + (size_t)n * F;
  float acc = 0.f;
  for (int c = 0; c < F; ++c) acc += r[c] * Wout[c];
  logits[n] = acc + bout[0];
}

// single-block global softmax over logits
__global__ void k_softmax(const float* __restrict__ logits, float* __restrict__ weights, int N) {
  __shared__ float red[1024];
  int t = threadIdx.x;
  float mx = -INFINITY;
  for (int i = t; i < N; i += 1024) mx = fmaxf(mx, logits[i]);
  red[t] = mx; __syncthreads();
  for (int s = 512; s > 0; s >>= 1) { if (t < s) red[t] = fmaxf(red[t], red[t + s]); __syncthreads(); }
  mx = red[0]; __syncthreads();
  float sum = 0.f;
  for (int i = t; i < N; i += 1024) sum += expf(logits[i] - mx);
  red[t] = sum; __syncthreads();
  for (int s = 512; s > 0; s >>= 1) { if (t < s) red[t] += red[t + s]; __syncthreads(); }
  float inv = 1.f / red[0];
  for (int i = t; i < N; i += 1024) weights[i] = expf(logits[i] - mx) * inv;
}

static inline int cdiv(long long a, long long b) { return (int)((a + b - 1) / b); }

extern "C" void kernel_launch(void* const* d_in, const int* in_sizes, int n_in,
                              void* d_out, int out_size, void* d_ws, size_t ws_size,
                              hipStream_t stream) {
  const float* x    = (const float*)d_in[0];
  const int*   eix  = (const int*)  d_in[1];
  const float* reg  = (const float*)d_in[2];
  const float* ng   = (const float*)d_in[3];
  const float* nb   = (const float*)d_in[4];
  const float* rg   = (const float*)d_in[5];
  const float* rb   = (const float*)d_in[6];
  const float* W1   = (const float*)d_in[7];
  const float* as1w = (const float*)d_in[8];
  const float* ad1w = (const float*)d_in[9];
  const float* b1   = (const float*)d_in[10];
  const float* W2   = (const float*)d_in[11];
  const float* as2w = (const float*)d_in[12];
  const float* ad2w = (const float*)d_in[13];
  const float* b2   = (const float*)d_in[14];
  const float* Wout = (const float*)d_in[15];
  const float* bout = (const float*)d_in[16];

  const int N  = in_sizes[0] / 128;
  const int E  = in_sizes[1] / 2;
  const int* esrc = eix;
  const int* edst = eix + E;

  const int H1 = 4, C = 64, HC1 = 256, HC2 = 64;
  const int KPAD1 = 160, KT1 = 5, NT1 = 16;
  const int K2 = 256,  KT2 = 8, NT2 = 4;
  const int Np16 = ((N + 15) / 16) * 16;
  const int nRT  = Np16 / 16;

  // ---- workspace layout (256B aligned) ----
  size_t off = 0;
  auto alloc = [&](size_t bytes) { size_t r = off; off = (off + bytes + 255) & ~(size_t)255; return r; };
  char* ws = (char*)d_ws;
  float*    rln   = (float*)   (ws + alloc(16 * 4));
  _Float16* h16   = (_Float16*)(ws + alloc((size_t)Np16 * KPAD1 * 2));
  _Float16* W1p   = (_Float16*)(ws + alloc((size_t)KT1 * NT1 * 512 * 2));
  float*    xw1   = (float*)   (ws + alloc((size_t)N * HC1 * 4));
  float*    as1   = (float*)   (ws + alloc((size_t)N * H1 * 4));
  float*    ad1   = (float*)   (ws + alloc((size_t)N * H1 * 4));
  float*    m1    = (float*)   (ws + alloc((size_t)N * H1 * 4));
  float*    dn1   = (float*)   (ws + alloc((size_t)N * H1 * 4));
  float*    al1   = (float*)   (ws + alloc((size_t)(E + N) * H1 * 4));
  float*    out1  = (float*)   (ws + alloc((size_t)N * HC1 * 4));
  _Float16* h1_16 = (_Float16*)(ws + alloc((size_t)Np16 * HC1 * 2));
  _Float16* W2p   = (_Float16*)(ws + alloc((size_t)KT2 * NT2 * 512 * 2));
  float*    xw2   = (float*)   (ws + alloc((size_t)N * HC2 * 4));
  float*    as2   = (float*)   (ws + alloc((size_t)N * 4));
  float*    ad2   = (float*)   (ws + alloc((size_t)N * 4));
  float*    m2    = (float*)   (ws + alloc((size_t)N * 4));
  float*    dn2   = (float*)   (ws + alloc((size_t)N * 4));
  float*    al2   = (float*)   (ws + alloc((size_t)(E + N) * 4));
  float*    out2  = (float*)   (ws + alloc((size_t)N * HC2 * 4));

  float* weights = (float*)d_out;
  float* logits  = (float*)d_out + N;

  // ---- stage 0: layernorms + concat ----
  k_reg_ln<<<1, 32, 0, stream>>>(reg, rg, rb, rln, 16);
  k_node_ln<<<cdiv((long long)N * 32, 256), 256, 0, stream>>>(x, ng, nb, rln, h16, N);

  // ---- stage 1: GEMM1 (WMMA) ----
  k_pack_w<<<cdiv((long long)KT1 * NT1 * 512, 256), 256, 0, stream>>>(W1, W1p, 144, KT1, NT1, HC1);
  k_gemm_wmma<<<cdiv((long long)nRT * NT1 * 32, 256), 256, 0, stream>>>(
      h16, W1p, xw1, nRT, N, KPAD1, KT1, NT1, HC1);
  k_alpha<<<cdiv((long long)N * H1, 256), 256, 0, stream>>>(xw1, as1w, ad1w, as1, ad1, N, H1, C);

  // ---- stage 1: edge softmax + aggregate ----
  long long EH1 = (long long)(E + N) * H1;
  k_fill<<<cdiv((long long)N * H1, 256), 256, 0, stream>>>(m1, -INFINITY, (long long)N * H1);
  k_fill<<<cdiv((long long)N * H1, 256), 256, 0, stream>>>(dn1, 0.f, (long long)N * H1);
  k_fill<<<cdiv((long long)N * HC1, 256), 256, 0, stream>>>(out1, 0.f, (long long)N * HC1);
  k_edge_max<<<cdiv(EH1, 256), 256, 0, stream>>>(esrc, edst, E, N, H1, as1, ad1, m1);
  k_edge_exp<<<cdiv(EH1, 256), 256, 0, stream>>>(esrc, edst, E, N, H1, as1, ad1, m1, dn1, al1);
  k_edge_msg<<<cdiv((long long)(E + N) * HC1, 256), 256, 0, stream>>>(
      esrc, edst, E, N, H1, C, xw1, al1, dn1, out1);
  k_bias_elu_f16<<<cdiv((long long)N * HC1, 256), 256, 0, stream>>>(
      out1, b1, h1_16, (long long)N * HC1, HC1);

  // ---- stage 2: GEMM2 (WMMA) ----
  k_pack_w<<<cdiv((long long)KT2 * NT2 * 512, 256), 256, 0, stream>>>(W2, W2p, K2, KT2, NT2, HC2);
  k_gemm_wmma<<<cdiv((long long)nRT * NT2 * 32, 256), 256, 0, stream>>>(
      h1_16, W2p, xw2, nRT, N, K2, KT2, NT2, HC2);
  k_alpha<<<cdiv((long long)N, 256), 256, 0, stream>>>(xw2, as2w, ad2w, as2, ad2, N, 1, C);

  // ---- stage 2: edge softmax + aggregate ----
  long long EH2 = (long long)(E + N);
  k_fill<<<cdiv((long long)N, 256), 256, 0, stream>>>(m2, -INFINITY, (long long)N);
  k_fill<<<cdiv((long long)N, 256), 256, 0, stream>>>(dn2, 0.f, (long long)N);
  k_fill<<<cdiv((long long)N * HC2, 256), 256, 0, stream>>>(out2, 0.f, (long long)N * HC2);
  k_edge_max<<<cdiv(EH2, 256), 256, 0, stream>>>(esrc, edst, E, N, 1, as2, ad2, m2);
  k_edge_exp<<<cdiv(EH2, 256), 256, 0, stream>>>(esrc, edst, E, N, 1, as2, ad2, m2, dn2, al2);
  k_edge_msg<<<cdiv((long long)(E + N) * HC2, 256), 256, 0, stream>>>(
      esrc, edst, E, N, 1, C, xw2, al2, dn2, out2);
  k_bias_elu_f32<<<cdiv((long long)N * HC2, 256), 256, 0, stream>>>(
      out2, b2, (long long)N * HC2, HC2);

  // ---- output head ----
  k_logits<<<cdiv((long long)N, 256), 256, 0, stream>>>(out2, Wout, bout, logits, N, C);
  k_softmax<<<1, 1024, 0, stream>>>(logits, weights, N);
}
